// Cross_Attention_54666343743581
// MI455X (gfx1250) — compile-verified, matrix operand order checked by reading, not run on hardware
//
#include <hip/hip_runtime.h>
#include <hip/hip_bf16.h>

// Problem constants (match reference)
#define BB   4
#define NN   4096
#define DD   512
#define HH   8      // heads
#define HKC  64     // per-head channels
#define OD   1024   // 2*D output channels
#define LN_EPS 1e-5f

typedef __attribute__((ext_vector_type(16))) _Float16 v16h;
typedef __attribute__((ext_vector_type(8)))  _Float16 v8h;
typedef __attribute__((ext_vector_type(8)))  float    v8f;
typedef __attribute__((ext_vector_type(4)))  int      v4i;

// ---------------------------------------------------------------------------
// Async global->LDS 16B copy (CDNA5), with guarded fallback to sync copy.
// Builtin signature (from clang diagnostic): (v4i AS1*, v4i AS3*, int, int).
// ---------------------------------------------------------------------------
#if __has_builtin(__builtin_amdgcn_global_load_async_to_lds_b128)
#define ASYNC_LDS 1
typedef __attribute__((address_space(1))) v4i* as1_v4i;
typedef __attribute__((address_space(3))) v4i* as3_v4i;
#else
#define ASYNC_LDS 0
#endif

__device__ __forceinline__ void cp16_lds(const _Float16* g, _Float16* l) {
#if ASYNC_LDS
  __builtin_amdgcn_global_load_async_to_lds_b128((as1_v4i)g, (as3_v4i)l, 0, 0);
#else
  *(v8h*)l = *(const v8h*)g;
#endif
}

__device__ __forceinline__ void wait_async_all() {
#if ASYNC_LDS
#if __has_builtin(__builtin_amdgcn_s_wait_asynccnt)
  __builtin_amdgcn_s_wait_asynccnt(0);
#else
  asm volatile("s_wait_asynccnt 0x0" ::: "memory");
#endif
#endif
}

// Load one 16x32 f16 WMMA operand row for this lane from a row pointer whose
// 32 K-values are contiguous. Per ISA layout: element e -> k = (e<8?0:16) + half*8 + (e&7).
__device__ __forceinline__ v16h ld_row16(const _Float16* row, int half) {
  v8h lo = *(const v8h*)(row + half * 8);
  v8h hi = *(const v8h*)(row + 16 + half * 8);
  v16h r;
#pragma unroll
  for (int i = 0; i < 8; ++i) { r[i] = lo[i]; r[i + 8] = hi[i]; }
  return r;
}

__device__ __forceinline__ v8f wmma16(v16h a, v16h b, v8f c) {
  return __builtin_amdgcn_wmma_f32_16x16x32_f16(
      /*neg_a=*/false, a, /*neg_b=*/false, b,
      /*c_mod=*/(short)0, c, /*reuse_a=*/false, /*reuse_b=*/false);
}

// ---------------------------------------------------------------------------
// K0: one-shot conv_w f32 -> f16 (1 MB, stays L2-resident for K4).
// ---------------------------------------------------------------------------
__global__ void k_cvtw(const float* __restrict__ W, _Float16* __restrict__ Wh) {
  size_t i = ((size_t)blockIdx.x * 256 + threadIdx.x) * 4;
#pragma unroll
  for (int j = 0; j < 4; ++j) Wh[i + j] = (_Float16)W[i + j];
}

// ---------------------------------------------------------------------------
// K1: per (b, channel) online softmax stats over the token axis (N=4096).
// Threads = channels -> every global load step is a fully coalesced 1KB line.
// ---------------------------------------------------------------------------
__global__ void k_keystats(const float* __restrict__ x2,
                           float* __restrict__ rowmax, float* __restrict__ rowrcp) {
  int c  = blockIdx.x * blockDim.x + threadIdx.x;   // 0 .. B*D-1
  int b  = c / DD;
  int ch = c % DD;
  const float* p = x2 + (size_t)b * NN * DD + ch;
  float m = -1e30f, s = 0.0f;
  for (int n = 0; n < NN; ++n) {
    float x  = p[(size_t)n * DD];
    float mn = fmaxf(m, x);
    s = s * __expf(m - mn) + __expf(x - mn);
    m = mn;
  }
  rowmax[c] = m;
  rowrcp[c] = 1.0f / s;
}

// ---------------------------------------------------------------------------
// K2: context[b,h] (64x64) = softmax_N(keys) @ V^T, K = 4096 via WMMA.
// Stage exp-scaled keys E and values V *transposed* ([chan][token]) in LDS so
// both WMMA operands are two contiguous 16B LDS reads per lane.
// Output stored TRANSPOSED: ctxT[b,h][v][k] (f16) for K3's B operand.
// ---------------------------------------------------------------------------
__global__ void k_context(const float* __restrict__ x1, const float* __restrict__ x2,
                          const float* __restrict__ rowmax, const float* __restrict__ rowrcp,
                          _Float16* __restrict__ ctxT) {
  __shared__ __align__(16) _Float16 ldsE[64][32];
  __shared__ __align__(16) _Float16 ldsV[64][32];
  int bh = blockIdx.x;              // B*H = 32 blocks
  int b = bh >> 3, h = bh & 7;
  int tid = threadIdx.x;
  int lane = tid & 31, wave = tid >> 5;
  int half = lane >> 4, l16 = lane & 15;

  const v8f vzero = {};
  v8f acc[2]; acc[0] = vzero; acc[1] = vzero;
  const int tile0 = wave, tile1 = wave + 8;   // 16 (kt,vt) tiles over 8 waves

  const size_t baseX = (size_t)b * NN * DD + (size_t)h * HKC;
  const int cbase = b * DD + h * HKC;

  for (int n0 = 0; n0 < NN; n0 += 32) {
    __syncthreads();
#pragma unroll
    for (int i = 0; i < 8; ++i) {             // 2048 elems per matrix, 8/thread
      int idx = tid + i * 256;
      int nl = idx >> 6, c = idx & 63;        // coalesced over c
      size_t g = baseX + (size_t)(n0 + nl) * DD + c;
      float e = __expf(x2[g] - rowmax[cbase + c]) * rowrcp[cbase + c];
      ldsE[c][nl] = (_Float16)e;              // transposed stage
      ldsV[c][nl] = (_Float16)x1[g];
    }
    __syncthreads();
    {
      int kt = tile0 >> 2, vt = tile0 & 3;
      v16h a = ld_row16(&ldsE[kt * 16 + l16][0], half);
      v16h bm = ld_row16(&ldsV[vt * 16 + l16][0], half);
      acc[0] = wmma16(a, bm, acc[0]);
    }
    {
      int kt = tile1 >> 2, vt = tile1 & 3;
      v16h a = ld_row16(&ldsE[kt * 16 + l16][0], half);
      v16h bm = ld_row16(&ldsV[vt * 16 + l16][0], half);
      acc[1] = wmma16(a, bm, acc[1]);
    }
  }
  // D layout: VGPR r -> row k = kt*16 + r + 8*half ; lane -> col v = vt*16 + l16
#pragma unroll
  for (int t = 0; t < 2; ++t) {
    int tt = t == 0 ? tile0 : tile1;
    int kt = tt >> 2, vt = tt & 3;
#pragma unroll
    for (int r = 0; r < 8; ++r) {
      int k = kt * 16 + r + 8 * half;
      int v = vt * 16 + l16;
      ctxT[(size_t)bh * (HKC * HKC) + (size_t)v * HKC + k] = (_Float16)acc[t][r];
    }
  }
}

// ---------------------------------------------------------------------------
// K3: queries = channel softmax (64/head) -> LDS f16; attended per head via
// two chained WMMA (K=64) against ctxT (global, L2-hot). Writes aggregated f16.
// Block = 32 tokens, 8 waves: wave -> (token group of 16, 2 heads).
// ---------------------------------------------------------------------------
__global__ void k_attend(const float* __restrict__ x2, const _Float16* __restrict__ ctxT,
                         _Float16* __restrict__ agg) {
  __shared__ __align__(16) _Float16 ldsQ[32][DD];   // 32 KB
  int blk = blockIdx.x;                 // B*N/32 = 512
  int b = blk / (NN / 32);
  int n0 = (blk % (NN / 32)) * 32;
  int tid = threadIdx.x;

  // --- query softmax: one (token, head) row of 64 channels per thread ---
  {
    int tl = tid & 31, hh = tid >> 5;
    const float* px = x2 + ((size_t)b * NN + n0 + tl) * DD + hh * HKC;
    float m = -1e30f, s = 0.0f;
    for (int i = 0; i < HKC; ++i) {
      float x = px[i];
      float mn = fmaxf(m, x);
      s = s * __expf(m - mn) + __expf(x - mn);
      m = mn;
    }
    float rs = 1.0f / s;
    for (int i = 0; i < HKC; ++i)
      ldsQ[tl][hh * HKC + i] = (_Float16)(__expf(px[i] - m) * rs);
  }
  __syncthreads();

  int lane = tid & 31, wave = tid >> 5;
  int half = lane >> 4, l16 = lane & 15;
  int tg = wave >> 2;                  // token group 0..1
  int hbase = (wave & 3) * 2;          // two heads per wave
  const v8f vzero = {};

#pragma unroll
  for (int hj = 0; hj < 2; ++hj) {
    int h = hbase + hj;
    const _Float16* qrow = &ldsQ[tg * 16 + l16][h * HKC];
    v16h a0 = ld_row16(qrow, half);          // k = 0..31
    v16h a1 = ld_row16(qrow + 32, half);     // k = 32..63
    const _Float16* cb = ctxT + ((size_t)(b * HH + h)) * (HKC * HKC);
#pragma unroll
    for (int vt = 0; vt < 4; ++vt) {
      const _Float16* brow = cb + (size_t)(vt * 16 + l16) * HKC;  // ctxT[v][k]
      v16h b0 = ld_row16(brow, half);
      v16h b1 = ld_row16(brow + 32, half);
      v8f c = vzero;
      c = wmma16(a0, b0, c);
      c = wmma16(a1, b1, c);
#pragma unroll
      for (int r = 0; r < 8; ++r) {
        int tok = n0 + tg * 16 + r + 8 * half;
        int d = h * HKC + vt * 16 + l16;
        agg[((size_t)b * NN + tok) * DD + d] = (_Float16)c[r];
      }
    }
  }
}

// ---------------------------------------------------------------------------
// K4: reproj GEMM [16 tokens x 1024 outs, K=512] + bias + LayerNorm(1024).
// Software-pipelined: output quarters (256 ch) x 16 K-steps; W chunk (16 KB)
// double-buffered and staged with async global->LDS b128 copies; A chunks
// (f16 agg) staged once into 16 buffers during the first quarter and reused.
// 8 f32 accumulator tiles/wave in VGPRs; LN via shfl_xor + tiny LDS reduce.
// ---------------------------------------------------------------------------
__global__ void k_reproj_ln(const _Float16* __restrict__ agg, const _Float16* __restrict__ Wh,
                            const float* __restrict__ bias, const float* __restrict__ lnw,
                            const float* __restrict__ lnb, float* __restrict__ out) {
  __shared__ __align__(16) _Float16 ldsW[2][256][32];   // 32 KB double-buffered
  __shared__ __align__(16) _Float16 ldsA[16][16][32];   // 16 KB: all K chunks of A
  __shared__ float red[16][8][2];
  __shared__ float stats[16][2];

  int blk = blockIdx.x;                 // B*N/16 = 1024
  int b = blk >> 8;                     // 256 blocks per batch
  int n0 = (blk & 255) * 16;
  int tid = threadIdx.x, lane = tid & 31, wave = tid >> 5;
  int half = lane >> 4, l16 = lane & 15;

  const v8f vzero = {};
  v8f acc[8];
#pragma unroll
  for (int j = 0; j < 8; ++j) acc[j] = vzero;

  const size_t arow = ((size_t)b * NN + n0) * DD;   // A tile base (16 rows x 512)

#pragma unroll
  for (int oq = 0; oq < 4; ++oq) {                  // output quarter: 256 channels
    // Prologue: stage K-step 0 of this quarter into buffer 0.
    wait_async_all();
    __syncthreads();                                // protect buf0 from prior readers
#pragma unroll
    for (int i = 0; i < 4; ++i) {                   // W: 1024 16B chunks / 256 thr
      int c = tid + i * 256;
      int o = c >> 2, part = c & 3;
      cp16_lds(&Wh[(size_t)(oq * 256 + o) * DD + part * 8], &ldsW[0][o][part * 8]);
    }
    if (oq == 0 && tid < 64) {                      // A chunk 0: 64 chunks
      int tl = tid >> 2, part = tid & 3;
      cp16_lds(&agg[arow + (size_t)tl * DD + part * 8], &ldsA[0][tl][part * 8]);
    }

    for (int ks = 0; ks < 16; ++ks) {
      int cur = ks & 1;
      wait_async_all();
      __syncthreads();                              // stage for this step landed
      if (ks < 15) {                                // issue next stage into other buf
        int k1 = (ks + 1) * 32;
#pragma unroll
        for (int i = 0; i < 4; ++i) {
          int c = tid + i * 256;
          int o = c >> 2, part = c & 3;
          cp16_lds(&Wh[(size_t)(oq * 256 + o) * DD + k1 + part * 8],
                   &ldsW[cur ^ 1][o][part * 8]);
        }
        if (oq == 0 && tid < 64) {
          int tl = tid >> 2, part = tid & 3;
          cp16_lds(&agg[arow + (size_t)tl * DD + k1 + part * 8],
                   &ldsA[ks + 1][tl][part * 8]);
        }
      }
      v16h a = ld_row16(&ldsA[ks][l16][0], half);
      {
        v16h bm = ld_row16(&ldsW[cur][(wave * 2 + 0) * 16 + l16][0], half);
        acc[oq * 2 + 0] = wmma16(a, bm, acc[oq * 2 + 0]);
      }
      {
        v16h bm = ld_row16(&ldsW[cur][(wave * 2 + 1) * 16 + l16][0], half);
        acc[oq * 2 + 1] = wmma16(a, bm, acc[oq * 2 + 1]);
      }
    }
  }

  // fold bias into accumulators; channel o = oq*256 + (wave*2+jw)*16 + l16
#pragma unroll
  for (int j = 0; j < 8; ++j) {
    int o = (j >> 1) * 256 + (wave * 2 + (j & 1)) * 16 + l16;
    float bv = bias[o];
#pragma unroll
    for (int r = 0; r < 8; ++r) acc[j][r] += bv;
  }

  // per-lane partial sums per token row r (token = r + 8*half)
  float s[8], s2[8];
#pragma unroll
  for (int r = 0; r < 8; ++r) { s[r] = 0.0f; s2[r] = 0.0f; }
#pragma unroll
  for (int j = 0; j < 8; ++j)
#pragma unroll
    for (int r = 0; r < 8; ++r) { float x = acc[j][r]; s[r] += x; s2[r] += x * x; }

  // reduce across the 16 column lanes of each half-group
#pragma unroll
  for (int m = 1; m < 16; m <<= 1)
#pragma unroll
    for (int r = 0; r < 8; ++r) {
      s[r]  += __shfl_xor(s[r],  m, 32);
      s2[r] += __shfl_xor(s2[r], m, 32);
    }
  if (l16 == 0) {
#pragma unroll
    for (int r = 0; r < 8; ++r) {
      int tok = r + 8 * half;
      red[tok][wave][0] = s[r];
      red[tok][wave][1] = s2[r];
    }
  }
  __syncthreads();
  if (tid < 16) {
    float ts = 0.0f, ts2 = 0.0f;
    for (int w = 0; w < 8; ++w) { ts += red[tid][w][0]; ts2 += red[tid][w][1]; }
    float mu  = ts  * (1.0f / (float)OD);
    float var = ts2 * (1.0f / (float)OD) - mu * mu;
    stats[tid][0] = mu;
    stats[tid][1] = rsqrtf(var + LN_EPS);
  }
  __syncthreads();

#pragma unroll
  for (int j = 0; j < 8; ++j) {
    int o = (j >> 1) * 256 + (wave * 2 + (j & 1)) * 16 + l16;
    float g = lnw[o], be = lnb[o];
#pragma unroll
    for (int r = 0; r < 8; ++r) {
      int tok = r + 8 * half;
      float mu = stats[tok][0], rs = stats[tok][1];
      out[((size_t)b * NN + n0 + tok) * OD + o] = (acc[j][r] - mu) * rs * g + be;
    }
  }
}

// ---------------------------------------------------------------------------
// Launch
// ---------------------------------------------------------------------------
extern "C" void kernel_launch(void* const* d_in, const int* in_sizes, int n_in,
                              void* d_out, int out_size, void* d_ws, size_t ws_size,
                              hipStream_t stream) {
  const float* x1     = (const float*)d_in[0];
  const float* x2     = (const float*)d_in[1];
  const float* conv_w = (const float*)d_in[2];
  const float* conv_b = (const float*)d_in[3];
  const float* ln_w   = (const float*)d_in[4];
  const float* ln_b   = (const float*)d_in[5];
  float* out = (float*)d_out;

  char* ws = (char*)d_ws;
  float*    rowmax = (float*)(ws);                         //    8 KB
  float*    rowrcp = (float*)(ws + 8192);                  //    8 KB
  _Float16* ctxT   = (_Float16*)(ws + 16384);              //  256 KB
  _Float16* Wh     = (_Float16*)(ws + 278528);             //    1 MB (f16 conv_w)
  _Float16* agg    = (_Float16*)(ws + 1327104);            // 16.8 MB

  k_cvtw     <<< (OD * DD) / 1024, 256, 0, stream >>>(conv_w, Wh);
  k_keystats <<< (BB * DD) / 256,  256, 0, stream >>>(x2, rowmax, rowrcp);
  k_context  <<< BB * HH,          256, 0, stream >>>(x1, x2, rowmax, rowrcp, ctxT);
  k_attend   <<< (BB * NN) / 32,   256, 0, stream >>>(x2, ctxT, agg);
  k_reproj_ln<<< (BB * NN) / 16,   256, 0, stream >>>(agg, Wh, conv_b, ln_w, ln_b, out);
}
